// CrossFusionModule_23819888623718
// MI455X (gfx1250) — matrix operand order repaired and checked by reading, not compiled, not run on hardware
//
#include <hip/hip_runtime.h>
#include <hip/hip_bf16.h>

// CrossFusionModule fully fused for MI455X (gfx1250).
// One workgroup (8 wave32) per batch element; all intermediates in LDS (252KB/320KB).
// All GEMMs via v_wmma_f32_16x16x32_bf16 with fp32 accumulation.
// Global->LDS staging is double-buffered through registers (b128 loads) so HBM
// latency overlaps the WMMA pipeline.

#define TB 256
#define NB 512   // batch
#define SS 64    // sequence
#define FF 768   // input feature
#define DD 256   // projected dim

typedef __attribute__((ext_vector_type(16))) __bf16        v16bf;
typedef __attribute__((ext_vector_type(8)))  float         v8f;
typedef __attribute__((ext_vector_type(4)))  unsigned int  u32x4;
typedef __attribute__((ext_vector_type(4)))  float         f32x4;

union FragBF { v16bf v; u32x4 q[2]; };

// ---------------- LDS arena layout (bytes) ----------------
constexpr int A_TILE_E  = SS * 32;                    // feat tile elems (bf16)
constexpr int W_TILE_E  = DD * 32;                    // weight tile elems (bf16)
constexpr int OFF_BIAS  = 0;                          // ba[256],bv[256],bb[64],g[64],be[64] f32
constexpr int OFF_A     = 3072;                       // bf16 a   [64][256]   32KB
constexpr int OFF_V     = OFF_A    + SS * DD * 2;     // bf16 v   [64][256]   32KB
constexpr int OFF_AT    = OFF_V    + SS * DD * 2;     // bf16 aT  [256][64]   32KB
constexpr int OFF_VT    = OFF_AT   + DD * SS * 2;     // bf16 vT  [256][64]   32KB
constexpr int OFF_WBT   = OFF_AT;                     // bf16 WbT [64][512]   64KB (overlays dead aT/vT)
constexpr int OFF_T     = OFF_VT   + DD * SS * 2;     // bf16 t   [64][256]   32KB
constexpr int OFF_CC    = OFF_T    + SS * DD * 2;     // f32  cc  [64][64]    16KB
constexpr int OFF_STGA  = OFF_CC   + SS * SS * 4;     // bf16 feat tile x2 [64][32]  8KB
constexpr int OFF_STGW  = OFF_STGA + 2 * A_TILE_E * 2;// bf16 W tile^T x2 [256][32] 32KB
constexpr int OFF_FUSED = OFF_T;                      // bf16 fused [64][512] 64KB (overlays dead t/cc/stg)
constexpr int OFF_STAT  = OFF_STGW + 2 * W_TILE_E * 2;// f32 [4][64] softmax / LN stats
constexpr int OFF_P     = OFF_STAT + 4 * SS * 4;      // bf16 audio_att^T [64][64] 8KB
constexpr int OFF_Q     = OFF_P    + SS * SS * 2;     // bf16 visual_att  [64][64] 8KB
constexpr int OFF_H     = OFF_Q    + SS * SS * 2;     // f32 h [64][64] 16KB
constexpr int SMEM_BYTES = OFF_H   + SS * SS * 4;     // = 258048 (252KB) <= 320KB WGP LDS

// ---------------- helpers ----------------
__device__ __forceinline__ unsigned short f2bf(float f) {
  unsigned int u = __float_as_uint(f);
  u += 0x7FFFu + ((u >> 16) & 1u);          // round-to-nearest-even
  return (unsigned short)(u >> 16);
}
__device__ __forceinline__ float bf2f(unsigned short h) {
  return __uint_as_float(((unsigned int)h) << 16);
}
__device__ __forceinline__ v8f zero8() {
  v8f z; for (int i = 0; i < 8; ++i) z[i] = 0.0f; return z;
}

// A fragment (16x32 bf16): lane<16 rows M=0..15 hold K 0-7 & 16-23; lane>=16 hold K 8-15 & 24-31.
__device__ __forceinline__ v16bf ldA(const unsigned short* base, int stride, int row, int k0, int lane) {
  const int kb = (lane & 16) ? 8 : 0;
  const unsigned short* p = base + row * stride + k0 + kb;
  FragBF f;
  f.q[0] = *(const u32x4*)(p);
  f.q[1] = *(const u32x4*)(p + 16);
  return f.v;
}
// B fragment (32x16 bf16): lane<16 col N=lane holds K 0-7 & 16-23; lane>=16 col N=lane-16 holds K 8-15 & 24-31.
// 'base' must hold B transposed: base[col*stride + k].
__device__ __forceinline__ v16bf ldB(const unsigned short* base, int stride, int col, int k0, int lane) {
  const int kb = (lane & 16) ? 8 : 0;
  const unsigned short* p = base + col * stride + k0 + kb;
  FragBF f;
  f.q[0] = *(const u32x4*)(p);
  f.q[1] = *(const u32x4*)(p + 16);
  return f.v;
}
__device__ __forceinline__ v8f wmma_bf16(v16bf a, v16bf b, v8f c) {
  return __builtin_amdgcn_wmma_f32_16x16x32_bf16(false, a, false, b, (short)0, c, false, false);
}

__global__ __launch_bounds__(TB, 1)
void cross_fusion_gfx1250(const float* __restrict__ audio, const float* __restrict__ visual,
                          const float* __restrict__ Wa,    const float* __restrict__ ba,
                          const float* __restrict__ Wv,    const float* __restrict__ bv,
                          const float* __restrict__ corr,  const float* __restrict__ Wb,
                          const float* __restrict__ bb,    const float* __restrict__ gammav,
                          const float* __restrict__ betav, float* __restrict__ out)
{
  extern __shared__ char smem[];
  const int tid  = threadIdx.x;
  const int lane = tid & 31;
  const int w    = tid >> 5;            // wave id, 0..7 (wave32)
  const int b    = blockIdx.x;

  unsigned short* sA    = (unsigned short*)(smem + OFF_A);
  unsigned short* sV    = (unsigned short*)(smem + OFF_V);
  unsigned short* sAT   = (unsigned short*)(smem + OFF_AT);
  unsigned short* sVT   = (unsigned short*)(smem + OFF_VT);
  unsigned short* sT    = (unsigned short*)(smem + OFF_T);
  float*          sCC   = (float*)         (smem + OFF_CC);
  unsigned short* sStgA = (unsigned short*)(smem + OFF_STGA);
  unsigned short* sStgW = (unsigned short*)(smem + OFF_STGW);
  unsigned short* sP    = (unsigned short*)(smem + OFF_P);
  unsigned short* sQ    = (unsigned short*)(smem + OFF_Q);
  float*          sStat = (float*)         (smem + OFF_STAT);
  float*          sH    = (float*)         (smem + OFF_H);

  float* sBa = (float*)(smem + OFF_BIAS);
  float* sBv = sBa + 256;
  float* sBb = sBv + 256;
  float* sGa = sBb + 64;
  float* sBe = sGa + 64;
  sBa[tid] = ba[tid];
  sBv[tid] = bv[tid];
  if (tid < 64) { sBb[tid] = bb[tid]; sGa[tid] = gammav[tid]; sBe[tid] = betav[tid]; }

  const int mt = w & 3;                 // M tile (4 x 16 rows)
  const int nh = w >> 2;                // N half for 256-wide GEMMs
  const int m0 = mt * 16;

  // staging work decomposition (per thread)
  const int fr  = tid >> 2, fcb = (tid & 3) * 8;   // feat tile: 8 contiguous f32
  const int wr  = tid >> 3, wcb = (tid & 7) * 32;  // weight tile: 32 contiguous f32 of one row

  // ======== Phase 1: projections  a = audio@Wa+ba, v = visual@Wv+bv  (M=64,N=256,K=768) ========
  for (int pr = 0; pr < 2; ++pr) {
    const float* feat = pr ? visual : audio;
    const float* W    = pr ? Wv : Wa;
    const float* bias = pr ? sBv : sBa;
    unsigned short* sO  = pr ? sV  : sA;
    unsigned short* sOT = pr ? sVT : sAT;

    v8f acc[8];
    for (int t = 0; t < 8; ++t) acc[t] = zero8();

    f32x4 freg0, freg1;          // feat stage regs (8 f32)
    f32x4 wreg[8];               // weight stage regs (32 f32)

    // prologue: load + commit tile 0 into buffer 0
    {
      const float* fp = feat + ((size_t)b * SS + fr) * FF + fcb;
      freg0 = *(const f32x4*)(fp);
      freg1 = *(const f32x4*)(fp + 4);
      const float* wp = W + (size_t)wr * DD + wcb;
      for (int j = 0; j < 8; ++j) wreg[j] = *(const f32x4*)(wp + 4 * j);

      unsigned short* sa = sStgA + fr * 32 + fcb;
      sa[0]=f2bf(freg0[0]); sa[1]=f2bf(freg0[1]); sa[2]=f2bf(freg0[2]); sa[3]=f2bf(freg0[3]);
      sa[4]=f2bf(freg1[0]); sa[5]=f2bf(freg1[1]); sa[6]=f2bf(freg1[2]); sa[7]=f2bf(freg1[3]);
      for (int j = 0; j < 8; ++j)
        for (int e = 0; e < 4; ++e)
          sStgW[(wcb + j * 4 + e) * 32 + wr] = f2bf(wreg[j][e]);   // transposed: [n][k]
    }
    __syncthreads();

    for (int k0 = 0, it = 0; k0 < FF; k0 += 32, ++it) {
      const int  cur  = it & 1;
      const bool more = (k0 + 32) < FF;
      if (more) { // issue global loads of next tile (overlaps the WMMAs below)
        const float* fp = feat + ((size_t)b * SS + fr) * FF + (k0 + 32) + fcb;
        freg0 = *(const f32x4*)(fp);
        freg1 = *(const f32x4*)(fp + 4);
        const float* wp = W + (size_t)(k0 + 32 + wr) * DD + wcb;
        for (int j = 0; j < 8; ++j) wreg[j] = *(const f32x4*)(wp + 4 * j);
      }
      const unsigned short* stgA = sStgA + cur * A_TILE_E;
      const unsigned short* stgW = sStgW + cur * W_TILE_E;
      v16bf af = ldA(stgA, 32, m0 + (lane & 15), 0, lane);
      for (int t = 0; t < 8; ++t) {
        const int n0 = nh * 128 + t * 16;
        v16bf bfv = ldB(stgW, 32, n0 + (lane & 15), 0, lane);
        acc[t] = wmma_bf16(af, bfv, acc[t]);
      }
      __syncthreads();                 // all waves done reading buf[1-cur] (prev iter) & buf[cur]
      if (more) {                      // commit next tile into the other buffer
        unsigned short* sa = sStgA + (1 - cur) * A_TILE_E + fr * 32 + fcb;
        sa[0]=f2bf(freg0[0]); sa[1]=f2bf(freg0[1]); sa[2]=f2bf(freg0[2]); sa[3]=f2bf(freg0[3]);
        sa[4]=f2bf(freg1[0]); sa[5]=f2bf(freg1[1]); sa[6]=f2bf(freg1[2]); sa[7]=f2bf(freg1[3]);
        unsigned short* swp = sStgW + (1 - cur) * W_TILE_E;
        for (int j = 0; j < 8; ++j)
          for (int e = 0; e < 4; ++e)
            swp[(wcb + j * 4 + e) * 32 + wr] = f2bf(wreg[j][e]);
      }
      __syncthreads();
    }
    // writeout with bias, both row-major and transposed (C layout: lane->N=lane%16, M=r+8*(lane/16))
    {
      const int mb = m0 + ((lane >> 4) << 3);
      for (int t = 0; t < 8; ++t) {
        const int n = nh * 128 + t * 16 + (lane & 15);
        const float bsv = bias[n];
        for (int r = 0; r < 8; ++r) {
          const int m = mb + r;
          const unsigned short h = f2bf(acc[t][r] + bsv);
          sO[m * DD + n]  = h;
          sOT[n * SS + m] = h;
        }
      }
    }
    __syncthreads();
  }

  // ======== Phase 2: t = a @ corr  (M=64,N=256,K=256), corr tiles double-buffered ========
  {
    v8f acc[8];
    for (int t = 0; t < 8; ++t) acc[t] = zero8();
    f32x4 wreg[8];
    { // prologue: corr tile 0
      const float* wp = corr + (size_t)wr * DD + wcb;
      for (int j = 0; j < 8; ++j) wreg[j] = *(const f32x4*)(wp + 4 * j);
      for (int j = 0; j < 8; ++j)
        for (int e = 0; e < 4; ++e)
          sStgW[(wcb + j * 4 + e) * 32 + wr] = f2bf(wreg[j][e]);
    }
    __syncthreads();
    for (int k0 = 0, it = 0; k0 < DD; k0 += 32, ++it) {
      const int  cur  = it & 1;
      const bool more = (k0 + 32) < DD;
      if (more) {
        const float* wp = corr + (size_t)(k0 + 32 + wr) * DD + wcb;
        for (int j = 0; j < 8; ++j) wreg[j] = *(const f32x4*)(wp + 4 * j);
      }
      const unsigned short* stgW = sStgW + cur * W_TILE_E;
      v16bf af = ldA(sA, DD, m0 + (lane & 15), k0, lane);
      for (int t = 0; t < 8; ++t) {
        const int n0 = nh * 128 + t * 16;
        v16bf bfv = ldB(stgW, 32, n0 + (lane & 15), 0, lane);
        acc[t] = wmma_bf16(af, bfv, acc[t]);
      }
      __syncthreads();
      if (more) {
        unsigned short* swp = sStgW + (1 - cur) * W_TILE_E;
        for (int j = 0; j < 8; ++j)
          for (int e = 0; e < 4; ++e)
            swp[(wcb + j * 4 + e) * 32 + wr] = f2bf(wreg[j][e]);
      }
      __syncthreads();
    }
    const int mb = m0 + ((lane >> 4) << 3);
    for (int t = 0; t < 8; ++t) {
      const int n = nh * 128 + t * 16 + (lane & 15);
      for (int r = 0; r < 8; ++r) sT[(mb + r) * DD + n] = f2bf(acc[t][r]);
    }
    __syncthreads();
  }

  // ======== Phase 3: cc = t @ v^T  (M=64,N=64,K=256; B col n = v row n, contiguous) ========
  {
    const int ntb = (w >> 2) * 2;
    v8f acc0 = zero8(), acc1 = zero8();
    for (int k0 = 0; k0 < DD; k0 += 32) {
      v16bf af = ldA(sT, DD, m0 + (lane & 15), k0, lane);
      v16bf b0 = ldB(sV, DD, (ntb + 0) * 16 + (lane & 15), k0, lane);
      v16bf b1 = ldB(sV, DD, (ntb + 1) * 16 + (lane & 15), k0, lane);
      acc0 = wmma_bf16(af, b0, acc0);
      acc1 = wmma_bf16(af, b1, acc1);
    }
    const int mb = m0 + ((lane >> 4) << 3);
    const int n0 = (ntb + 0) * 16 + (lane & 15);
    const int n1 = (ntb + 1) * 16 + (lane & 15);
    for (int r = 0; r < 8; ++r) {
      sCC[(mb + r) * SS + n0] = acc0[r];
      sCC[(mb + r) * SS + n1] = acc1[r];
    }
    __syncthreads();
  }

  // ======== Phase 4: softmax stats + P = audio_att^T, Q = visual_att (bf16) ========
  {
    float* rmax = sStat;        // row stats (softmax over j)
    float* rinv = sStat + 64;
    float* cmax = sStat + 128;  // col stats (softmax over i)
    float* cinv = sStat + 192;
    if (tid < 64) {
      float mx = -3.402823466e38f;
      for (int j = 0; j < SS; ++j) mx = fmaxf(mx, sCC[tid * SS + j]);
      float s = 0.0f;
      for (int j = 0; j < SS; ++j) s += __expf(sCC[tid * SS + j] - mx);
      rmax[tid] = mx; rinv[tid] = 1.0f / s;
    } else if (tid < 128) {
      const int c = tid - 64;
      float mx = -3.402823466e38f;
      for (int i = 0; i < SS; ++i) mx = fmaxf(mx, sCC[i * SS + c]);
      float s = 0.0f;
      for (int i = 0; i < SS; ++i) s += __expf(sCC[i * SS + c] - mx);
      cmax[c] = mx; cinv[c] = 1.0f / s;
    }
    __syncthreads();
    for (int idx = tid; idx < SS * SS; idx += TB) {
      const int i = idx >> 6, j = idx & 63;
      const float cval = sCC[idx];
      sQ[idx]          = f2bf(__expf(cval - rmax[i]) * rinv[i]);  // visual_att[i][j]
      sP[j * SS + i]   = f2bf(__expf(cval - cmax[j]) * cinv[j]);  // audio_att^T[j][i]
    }
    __syncthreads();
  }

  // ======== Phase 5: fused[:,0:256]   = P @ a + a   (att_a^T)
  //                   fused[:,256:512] = Q @ v + v   (att_v^T)   (M=64,N=256,K=64) ========
  {
    unsigned short* sF = (unsigned short*)(smem + OFF_FUSED);
    const int g = w >> 2; // 0 -> audio branch, 1 -> visual branch
    const unsigned short* Pm  = g ? sQ  : sP;
    const unsigned short* Bt  = g ? sVT : sAT;   // B col d = (a|v)T row d, contiguous
    const unsigned short* Res = g ? sV  : sA;
    for (int nt = 0; nt < 16; ++nt) {
      const int n0 = nt * 16;
      v8f acc = zero8();
      for (int k0 = 0; k0 < SS; k0 += 32) {
        v16bf af  = ldA(Pm, SS, m0 + (lane & 15), k0, lane);
        v16bf bfv = ldB(Bt, SS, n0 + (lane & 15), k0, lane);
        acc = wmma_bf16(af, bfv, acc);
      }
      const int n  = n0 + (lane & 15);
      const int mb = m0 + ((lane >> 4) << 3);
      for (int r = 0; r < 8; ++r) {
        const int m = mb + r;
        sF[m * 512 + g * DD + n] = f2bf(acc[r] + bf2f(Res[m * DD + n]));
      }
    }
    __syncthreads();
  }

  // ======== Phase 6: h = fused @ Wb + bb  (M=64,N=64,K=512) ========
  {
    unsigned short* sWbT = (unsigned short*)(smem + OFF_WBT); // overlays dead aT/vT
    unsigned short* sF   = (unsigned short*)(smem + OFF_FUSED);
    // stage Wb [512][64] transposed -> sWbT[n][k] with b128 loads (32 f32x4 / thread)
    for (int i = 0; i < 32; ++i) {
      const int idx  = tid + i * TB;       // 0..8191 quads
      const int k    = idx >> 4;
      const int cgrp = idx & 15;
      f32x4 q = *(const f32x4*)(Wb + (size_t)k * 64 + cgrp * 4);
      for (int e = 0; e < 4; ++e) sWbT[(cgrp * 4 + e) * 512 + k] = f2bf(q[e]);
    }
    __syncthreads();
    const int ntb = (w >> 2) * 2;
    v8f acc0 = zero8(), acc1 = zero8();
    for (int k0 = 0; k0 < 512; k0 += 32) {
      v16bf af = ldA(sF, 512, m0 + (lane & 15), k0, lane);
      v16bf b0 = ldB(sWbT, 512, (ntb + 0) * 16 + (lane & 15), k0, lane);
      v16bf b1 = ldB(sWbT, 512, (ntb + 1) * 16 + (lane & 15), k0, lane);
      acc0 = wmma_bf16(af, b0, acc0);
      acc1 = wmma_bf16(af, b1, acc1);
    }
    const int mb = m0 + ((lane >> 4) << 3);
    const int n0 = (ntb + 0) * 16 + (lane & 15);
    const int n1 = (ntb + 1) * 16 + (lane & 15);
    for (int r = 0; r < 8; ++r) {
      sH[(mb + r) * SS + n0] = acc0[r] + sBb[n0];
      sH[(mb + r) * SS + n1] = acc1[r] + sBb[n1];
    }
    __syncthreads();
  }

  // ======== Phase 7: LayerNorm(64) * gamma + beta, ReLU, store ========
  {
    float* mu   = sStat;       // reuse stats region
    float* rstd = sStat + 64;
    if (tid < 64) {
      float s = 0.0f;
      for (int o = 0; o < 64; ++o) s += sH[tid * 64 + o];
      const float m = s * (1.0f / 64.0f);
      float v2 = 0.0f;
      for (int o = 0; o < 64; ++o) { const float d = sH[tid * 64 + o] - m; v2 += d * d; }
      mu[tid]   = m;
      rstd[tid] = rsqrtf(v2 * (1.0f / 64.0f) + 1e-5f);
    }
    __syncthreads();
    float* op = out + (size_t)b * (SS * 64);
    for (int idx = tid; idx < SS * 64; idx += TB) {
      const int row = idx >> 6, o = idx & 63;
      const float val = (sH[idx] - mu[row]) * rstd[row] * sGa[o] + sBe[o];
      op[idx] = fmaxf(val, 0.0f);
    }
  }
}

extern "C" void kernel_launch(void* const* d_in, const int* in_sizes, int n_in,
                              void* d_out, int out_size, void* d_ws, size_t ws_size,
                              hipStream_t stream) {
  const float* audio  = (const float*)d_in[0];
  const float* visual = (const float*)d_in[1];
  const float* Wa     = (const float*)d_in[2];
  const float* ba     = (const float*)d_in[3];
  const float* Wv     = (const float*)d_in[4];
  const float* bv     = (const float*)d_in[5];
  const float* corr   = (const float*)d_in[6];
  const float* Wb     = (const float*)d_in[7];
  const float* bb     = (const float*)d_in[8];
  const float* gam    = (const float*)d_in[9];
  const float* bet    = (const float*)d_in[10];
  float* out = (float*)d_out;

  (void)in_sizes; (void)n_in; (void)out_size; (void)d_ws; (void)ws_size;

  // CDNA5 WGP allows up to 320KB LDS per workgroup; raise the dynamic-LDS cap.
  hipFuncSetAttribute(reinterpret_cast<const void*>(cross_fusion_gfx1250),
                      hipFuncAttributeMaxDynamicSharedMemorySize, SMEM_BYTES);

  cross_fusion_gfx1250<<<NB, TB, SMEM_BYTES, stream>>>(
      audio, visual, Wa, ba, Wv, bv, corr, Wb, bb, gam, bet, out);
}
